// GNS_TAT_62251255989046
// MI455X (gfx1250) — matrix operand order repaired and checked by reading, not compiled, not run on hardware
//
#include <hip/hip_runtime.h>

// ---------------------------------------------------------------------------
// Problem constants (from the reference)
// ---------------------------------------------------------------------------
#define NN   50000      // nodes
#define EE   400000     // edges
#define FF   128        // features
#define HH   8          // heads (HD = 16)
#define RBF_ 100        // radial basis
#define F3   384        // 3*F
#define GK   128        // padded contraction dim for every WMMA GEMM

typedef __attribute__((ext_vector_type(16))) __bf16 v16bf;
typedef __attribute__((ext_vector_type(8)))  __bf16 v8bf;
typedef __attribute__((ext_vector_type(4)))  __bf16 v4bf;
typedef __attribute__((ext_vector_type(8)))  float  v8f;

__device__ __forceinline__ float silu_f(float x) { return x / (1.0f + expf(-x)); }

// ---------------------------------------------------------------------------
// Utility kernels
// ---------------------------------------------------------------------------
__global__ __launch_bounds__(256) void zero_f32(float* __restrict__ p, long long n) {
    long long t = blockIdx.x * (long long)blockDim.x + threadIdx.x;
    if (t < n) p[t] = 0.0f;
}

// fp32 -> bf16 cast with optional K-padding (kin < kpad rows are zero-filled)
__global__ __launch_bounds__(256) void cast_pad_bf16(const float* __restrict__ src,
                                                     __bf16* __restrict__ dst,
                                                     long long total, int kin, int kpad) {
    long long t = blockIdx.x * (long long)blockDim.x + threadIdx.x;
    if (t >= total) return;
    if (kin == kpad) { dst[t] = (__bf16)src[t]; return; }
    long long r = t / kpad;
    int c = (int)(t - r * kpad);
    dst[t] = (c < kin) ? (__bf16)src[r * kin + c] : (__bf16)0.0f;
}

// ---------------------------------------------------------------------------
// LayerNorm: one wave per node row (128 floats, 4/lane), bf16 output
// ---------------------------------------------------------------------------
__global__ __launch_bounds__(256) void layernorm_kernel(const float* __restrict__ x,
                                                        const float* __restrict__ w,
                                                        const float* __restrict__ b,
                                                        __bf16* __restrict__ xn) {
    const int wave = (blockIdx.x * blockDim.x + threadIdx.x) >> 5;
    if (wave >= NN) return;
    const int lane = threadIdx.x & 31;
    float4 xv = *(const float4*)(x + (size_t)wave * FF + lane * 4);

    float s = xv.x + xv.y + xv.z + xv.w;
    for (int m = 16; m; m >>= 1) s += __shfl_xor(s, m, 32);
    const float mu = s * (1.0f / FF);

    const float a0 = xv.x - mu, a1 = xv.y - mu, a2 = xv.z - mu, a3 = xv.w - mu;
    float v = a0 * a0 + a1 * a1 + a2 * a2 + a3 * a3;
    for (int m = 16; m; m >>= 1) v += __shfl_xor(v, m, 32);
    const float inv = rsqrtf(v * (1.0f / FF) + 1e-5f);

    float4 wv = *(const float4*)(w + lane * 4);
    float4 bv = *(const float4*)(b + lane * 4);
    v4bf o;
    o[0] = (__bf16)(a0 * inv * wv.x + bv.x);
    o[1] = (__bf16)(a1 * inv * wv.y + bv.y);
    o[2] = (__bf16)(a2 * inv * wv.z + bv.z);
    o[3] = (__bf16)(a3 * inv * wv.w + bv.w);
    *(v4bf*)(xn + (size_t)wave * FF + lane * 4) = o;
}

// ---------------------------------------------------------------------------
// WMMA GEMM:  Y(M x Nout) = A(M x 128, bf16 row-major) @ W(Nout x 128, bf16)^T + bias
// One wave owns one 16-row block of A; A fragments (K=128 -> 4) loaded once and
// reused across all Nout/16 column tiles. A/B fragments use a contiguous K order
// shared by A and B (not the ISA table order); C follows the f32 16x16 VGPR map.
//   A lane(l): row = tileM*16 + (l&15); elems j -> K = kb + 16g + j
//                                                          (g = l>>4)
//   B lane(l): col = tileN*16 + (l&15); elems j -> K = kb + 16g + j
//   C lane(l): col = (l&15); elem r -> row = r + 8g
// ---------------------------------------------------------------------------
template <bool SILU, bool BF16OUT>
__global__ __launch_bounds__(256) void gemm_wmma(const __bf16* __restrict__ A,
                                                 const __bf16* __restrict__ W,
                                                 const float* __restrict__ bias,
                                                 void* __restrict__ Y,
                                                 int M, int Nout) {
    const int lane = threadIdx.x & 31;
    const int wave = (blockIdx.x * blockDim.x + threadIdx.x) >> 5;
    const int Mtiles = M >> 4;
    if (wave >= Mtiles) return;
    const int l15 = lane & 15;
    const int g   = lane >> 4;
    const int arow = (wave << 4) + l15;

    v16bf afrag[GK / 32];
    const __bf16* arp = A + (size_t)arow * GK;
#pragma unroll
    for (int kt = 0; kt < GK / 32; ++kt) {
        const __bf16* p = arp + kt * 32 + 16 * g;
        v8bf a0 = *(const v8bf*)(p);
        v8bf a1 = *(const v8bf*)(p + 8);
#pragma unroll
        for (int i = 0; i < 8; ++i) { afrag[kt][i] = a0[i]; afrag[kt][i + 8] = a1[i]; }
    }

    const int Ntiles = Nout >> 4;
    for (int nt = 0; nt < Ntiles; ++nt) {
        const int bcol = (nt << 4) + l15;
        const float bval = bias[bcol];
        v8f acc;
#pragma unroll
        for (int i = 0; i < 8; ++i) acc[i] = bval;

        const __bf16* wrp = W + (size_t)bcol * GK + 16 * g;
#pragma unroll
        for (int kt = 0; kt < GK / 32; ++kt) {
            v16bf bfrag = *(const v16bf*)(wrp + kt * 32);
            acc = __builtin_amdgcn_wmma_f32_16x16x32_bf16(
                false, afrag[kt], false, bfrag, (short)0, acc, false, false);
        }

#pragma unroll
        for (int r = 0; r < 8; ++r) {
            const int m = (wave << 4) + r + 8 * g;
            float val = acc[r];
            if (SILU) val = silu_f(val);
            const size_t idx = (size_t)m * Nout + bcol;
            if (BF16OUT) ((__bf16*)Y)[idx] = (__bf16)val;
            else         ((float*)Y)[idx]  = val;
        }
    }
}

// ---------------------------------------------------------------------------
// vec_dot[n,f] = sum_c vp[n,c,f]*vp[n,c,128+f] ; vec3[n,c,f] = vp[n,c,256+f]
// ---------------------------------------------------------------------------
__global__ __launch_bounds__(256) void vecdot_vec3_kernel(const float* __restrict__ vp,
                                                          float* __restrict__ vdot,
                                                          float* __restrict__ vec3) {
    const int t = blockIdx.x * blockDim.x + threadIdx.x;   // N*128 threads
    if (t >= NN * FF) return;
    const int n = t >> 7, f = t & (FF - 1);
    float s = 0.0f;
#pragma unroll
    for (int c = 0; c < 3; ++c) {
        const size_t bse = (size_t)(n * 3 + c) * F3 + f;
        s += vp[bse] * vp[bse + FF];
        vec3[(size_t)(n * 3 + c) * FF + f] = vp[bse + 2 * FF];
    }
    vdot[t] = s;
}

// ---------------------------------------------------------------------------
// Edge message+scatter kernel: one wave32 per edge.
//   attn_h = silu(sum_d q[dst,h,d]*k[src,h,d]*dk[e,h,d]) * cutoff(r)
//   v_j = v[src]*dv[e]  -> (xm | v1m | v2m) per head (48 channels/head)
//   x_agg[dst]  += xm*attn          (atomic)
//   vec_agg[dst,c] += vec[src,c]*v1m + v2m*d_ij[c]   (atomic, into d_out dvec)
// Lane owns 4 channels f = lane*4..lane*4+3 (all within head h = lane>>2).
// ---------------------------------------------------------------------------
__global__ __launch_bounds__(256) void edge_kernel(const int* __restrict__ ei,
                                                   const float* __restrict__ rij,
                                                   const float* __restrict__ dij,
                                                   const float* __restrict__ q,
                                                   const float* __restrict__ k,
                                                   const float* __restrict__ v,
                                                   const __bf16* __restrict__ dk,
                                                   const __bf16* __restrict__ dv,
                                                   const float* __restrict__ vecin,
                                                   float* __restrict__ xagg,
                                                   float* __restrict__ vecagg) {
    const int wave = (blockIdx.x * blockDim.x + threadIdx.x) >> 5;
    if (wave >= EE) return;
    const int lane = threadIdx.x & 31;
    const int src = ei[wave];
    const int dst = ei[EE + wave];

    const float r = rij[wave];
    const float cut = (r < 6.0f) ? 0.5f * (cosf(r * 0.5235987755982988f) + 1.0f) : 0.0f;
    const float dc0 = dij[3 * (size_t)wave + 0];
    const float dc1 = dij[3 * (size_t)wave + 1];
    const float dc2 = dij[3 * (size_t)wave + 2];

    const int f0 = lane * 4;
    const float4 qv = *(const float4*)(q + (size_t)dst * FF + f0);
    const float4 kv = *(const float4*)(k + (size_t)src * FF + f0);
    const v4bf  dkb = *(const v4bf*)(dk + (size_t)wave * FF + f0);

    float s = qv.x * kv.x * (float)dkb[0] + qv.y * kv.y * (float)dkb[1] +
              qv.z * kv.z * (float)dkb[2] + qv.w * kv.w * (float)dkb[3];
    s += __shfl_xor(s, 1, 32);          // reduce over the 4 lanes of one head
    s += __shfl_xor(s, 2, 32);
    const float attn = silu_f(s) * cut;

    const int h = lane >> 2;            // head index
    const int d = f0 & 15;              // in-head channel base
    const size_t vb  = (size_t)src * F3 + h * 48 + d;
    const size_t dvb = (size_t)wave * F3 + h * 48 + d;
    const float4 vx = *(const float4*)(v + vb);
    const float4 v1 = *(const float4*)(v + vb + 16);
    const float4 v2 = *(const float4*)(v + vb + 32);
    const v4bf dvx = *(const v4bf*)(dv + dvb);
    const v4bf dv1 = *(const v4bf*)(dv + dvb + 16);
    const v4bf dv2 = *(const v4bf*)(dv + dvb + 32);

    // xm * attn -> x_agg
    float* xb = xagg + (size_t)dst * FF + f0;
    atomicAdd(xb + 0, vx.x * (float)dvx[0] * attn);
    atomicAdd(xb + 1, vx.y * (float)dvx[1] * attn);
    atomicAdd(xb + 2, vx.z * (float)dvx[2] * attn);
    atomicAdd(xb + 3, vx.w * (float)dvx[3] * attn);

    const float w1[4] = { v1.x * (float)dv1[0], v1.y * (float)dv1[1],
                          v1.z * (float)dv1[2], v1.w * (float)dv1[3] };
    const float w2[4] = { v2.x * (float)dv2[0], v2.y * (float)dv2[1],
                          v2.z * (float)dv2[2], v2.w * (float)dv2[3] };
    const float dc[3] = { dc0, dc1, dc2 };
#pragma unroll
    for (int c = 0; c < 3; ++c) {
        const float4 vv = *(const float4*)(vecin + (size_t)src * F3 + c * FF + f0);
        float* vg = vecagg + (size_t)dst * F3 + c * FF + f0;
        atomicAdd(vg + 0, vv.x * w1[0] + w2[0] * dc[c]);
        atomicAdd(vg + 1, vv.y * w1[1] + w2[1] * dc[c]);
        atomicAdd(vg + 2, vv.z * w1[2] + w2[2] * dc[c]);
        atomicAdd(vg + 3, vv.w * w1[3] + w2[3] * dc[c]);
    }
}

// ---------------------------------------------------------------------------
// Finalize: dx = vec_dot*o2 + o3 ; dvec (already = vec_agg) += vec3*o1
// ---------------------------------------------------------------------------
__global__ __launch_bounds__(256) void finalize_kernel(const float* __restrict__ o,
                                                       const float* __restrict__ vdot,
                                                       const float* __restrict__ vec3,
                                                       float* __restrict__ dx,
                                                       float* __restrict__ dvec) {
    const int t = blockIdx.x * blockDim.x + threadIdx.x;   // N*128 threads
    if (t >= NN * FF) return;
    const int n = t >> 7, f = t & (FF - 1);
    const float o1 = o[(size_t)n * F3 + f];
    const float o2 = o[(size_t)n * F3 + FF + f];
    const float o3 = o[(size_t)n * F3 + 2 * FF + f];
    dx[t] = vdot[t] * o2 + o3;
#pragma unroll
    for (int c = 0; c < 3; ++c) {
        const size_t i = (size_t)n * F3 + c * FF + f;
        dvec[i] += vec3[(size_t)(n * 3 + c) * FF + f] * o1;
    }
}

// ---------------------------------------------------------------------------
// Launch
// ---------------------------------------------------------------------------
extern "C" void kernel_launch(void* const* d_in, const int* in_sizes, int n_in,
                              void* d_out, int out_size, void* d_ws, size_t ws_size,
                              hipStream_t stream) {
    const float* x    = (const float*)d_in[0];
    const float* vec  = (const float*)d_in[1];
    const int*   ei   = (const int*)  d_in[2];
    const float* rij  = (const float*)d_in[3];
    const float* fij  = (const float*)d_in[4];
    const float* dij  = (const float*)d_in[5];
    const float* lnw  = (const float*)d_in[6];
    const float* lnb  = (const float*)d_in[7];
    const float* Wq   = (const float*)d_in[8];
    const float* bq   = (const float*)d_in[9];
    const float* Wk   = (const float*)d_in[10];
    const float* bk   = (const float*)d_in[11];
    const float* Wv   = (const float*)d_in[12];
    const float* bv   = (const float*)d_in[13];
    const float* Wvec = (const float*)d_in[14];
    const float* bvec = (const float*)d_in[15];
    const float* Wo   = (const float*)d_in[16];
    const float* bo   = (const float*)d_in[17];
    const float* Wdk  = (const float*)d_in[18];
    const float* bdk  = (const float*)d_in[19];
    const float* Wdv  = (const float*)d_in[20];
    const float* bdv  = (const float*)d_in[21];

    float* dx   = (float*)d_out;                       // N*F
    float* dvec = dx + (size_t)NN * FF;                // N*3*F (accumulates vec_agg)

    // ---- carve workspace (~910 MB total; vp/dv share one union region) ----
    char* base = (char*)d_ws;
    size_t off = 0;
    auto carve = [&](size_t bytes) -> void* {
        void* p = base + off;
        off += (bytes + 255) & ~(size_t)255;
        return p;
    };
    __bf16* xnb   = (__bf16*)carve((size_t)NN * FF * 2);
    float*  q     = (float*) carve((size_t)NN * FF * 4);
    float*  kk    = (float*) carve((size_t)NN * FF * 4);
    float*  v     = (float*) carve((size_t)NN * F3 * 4);
    __bf16* vecb  = (__bf16*)carve((size_t)NN * 3 * FF * 2);
    size_t vp_b = (size_t)NN * 3 * F3 * 4, dv_b = (size_t)EE * F3 * 2;
    void*   un    = carve(vp_b > dv_b ? vp_b : dv_b);  // vp (f32) then dv (bf16)
    float*  vdot  = (float*) carve((size_t)NN * FF * 4);
    float*  vec3  = (float*) carve((size_t)NN * 3 * FF * 4);
    __bf16* fijb  = (__bf16*)carve((size_t)EE * GK * 2);
    __bf16* dkb   = (__bf16*)carve((size_t)EE * FF * 2);
    float*  xagg  = (float*) carve((size_t)NN * FF * 4);
    __bf16* xaggb = (__bf16*)carve((size_t)NN * FF * 2);
    float*  o     = (float*) carve((size_t)NN * F3 * 4);
    __bf16* wqb   = (__bf16*)carve((size_t)FF * GK * 2);
    __bf16* wkb   = (__bf16*)carve((size_t)FF * GK * 2);
    __bf16* wvb   = (__bf16*)carve((size_t)F3 * GK * 2);
    __bf16* wvecb = (__bf16*)carve((size_t)F3 * GK * 2);
    __bf16* wob   = (__bf16*)carve((size_t)F3 * GK * 2);
    __bf16* wdkb  = (__bf16*)carve((size_t)FF * GK * 2);
    __bf16* wdvb  = (__bf16*)carve((size_t)F3 * GK * 2);
    (void)ws_size; (void)n_in; (void)in_sizes; (void)out_size;

    const dim3 B(256);
    auto blocks = [](long long total) { return (unsigned)((total + 255) / 256); };

    // 0) zero accumulators (xagg + dvec region of d_out)
    zero_f32<<<blocks((long long)NN * FF), B, 0, stream>>>(xagg, (long long)NN * FF);
    zero_f32<<<blocks((long long)NN * 3 * FF), B, 0, stream>>>(dvec, (long long)NN * 3 * FF);

    // 1) cast weights to bf16 (Wdk/Wdv are K-padded 100 -> 128)
    cast_pad_bf16<<<blocks((long long)FF * GK), B, 0, stream>>>(Wq, wqb, (long long)FF * GK, GK, GK);
    cast_pad_bf16<<<blocks((long long)FF * GK), B, 0, stream>>>(Wk, wkb, (long long)FF * GK, GK, GK);
    cast_pad_bf16<<<blocks((long long)F3 * GK), B, 0, stream>>>(Wv, wvb, (long long)F3 * GK, GK, GK);
    cast_pad_bf16<<<blocks((long long)F3 * GK), B, 0, stream>>>(Wvec, wvecb, (long long)F3 * GK, GK, GK);
    cast_pad_bf16<<<blocks((long long)F3 * GK), B, 0, stream>>>(Wo, wob, (long long)F3 * GK, GK, GK);
    cast_pad_bf16<<<blocks((long long)FF * GK), B, 0, stream>>>(Wdk, wdkb, (long long)FF * GK, RBF_, GK);
    cast_pad_bf16<<<blocks((long long)F3 * GK), B, 0, stream>>>(Wdv, wdvb, (long long)F3 * GK, RBF_, GK);

    // 2) layernorm -> xn (bf16); cast vec -> bf16; cast+pad f_ij -> bf16
    layernorm_kernel<<<NN / 8, B, 0, stream>>>(x, lnw, lnb, xnb);
    cast_pad_bf16<<<blocks((long long)NN * 3 * FF), B, 0, stream>>>(vec, vecb, (long long)NN * 3 * FF, GK, GK);
    cast_pad_bf16<<<blocks((long long)EE * GK), B, 0, stream>>>(fij, fijb, (long long)EE * GK, RBF_, GK);

    // 3) node GEMMs: q, k, v
    const unsigned gN  = (unsigned)((NN / 16 + 7) / 8);
    const unsigned g3N = (unsigned)((NN * 3 / 16 + 7) / 8);
    const unsigned gE  = (unsigned)((EE / 16 + 7) / 8);
    gemm_wmma<false, false><<<gN, B, 0, stream>>>(xnb, wqb, bq, q, NN, FF);
    gemm_wmma<false, false><<<gN, B, 0, stream>>>(xnb, wkb, bk, kk, NN, FF);
    gemm_wmma<false, false><<<gN, B, 0, stream>>>(xnb, wvb, bv, v, NN, F3);

    // 4) vp = vec @ Wvec^T (into union region), then vec_dot / vec3
    gemm_wmma<false, false><<<g3N, B, 0, stream>>>(vecb, wvecb, bvec, un, NN * 3, F3);
    vecdot_vec3_kernel<<<blocks((long long)NN * FF), B, 0, stream>>>((const float*)un, vdot, vec3);

    // 5) edge GEMMs with fused SiLU, bf16 outputs (dv reuses the vp region)
    gemm_wmma<true, true><<<gE, B, 0, stream>>>(fijb, wdkb, bdk, dkb, EE, FF);
    gemm_wmma<true, true><<<gE, B, 0, stream>>>(fijb, wdvb, bdv, un, EE, F3);

    // 6) edge message + scatter (one wave per edge)
    edge_kernel<<<EE / 8, B, 0, stream>>>(ei, rij, dij, q, kk, v, dkb,
                                          (const __bf16*)un, vec, xagg, dvec);

    // 7) o = x_agg @ Wo^T ; finalize outputs
    cast_pad_bf16<<<blocks((long long)NN * FF), B, 0, stream>>>(xagg, xaggb, (long long)NN * FF, GK, GK);
    gemm_wmma<false, false><<<gN, B, 0, stream>>>(xaggb, wob, bo, o, NN, F3);
    finalize_kernel<<<blocks((long long)NN * FF), B, 0, stream>>>(o, vdot, vec3, dx, dvec);
}
